// RNN_MultiRegional_49100066128404
// MI455X (gfx1250) — matrix operand order repaired and checked by reading, not compiled, not run on hardware
//
#include <hip/hip_runtime.h>
#include <hip/hip_fp16.h>

typedef _Float16 v8h  __attribute__((ext_vector_type(8)));
typedef _Float16 v16h __attribute__((ext_vector_type(16)));
typedef float    v8f  __attribute__((ext_vector_type(8)));

#define HIDD   512
#define HTOT   2560
#define B_     32
#define T_     1000
#define INP_   4
#define TAUF   0.1f
#define NEXC   359      // 512 - int(0.3*512)

#define NWG        40
#define BLK_SLICE  16384u             // bytes per (block, n-tile) weight slice (16x32x16 halves)
#define SLOT_LDS   49152u             // 3 blocks max per out-region
#define LDS_BYTES  (4u * SLOT_LDS)    // 192 KB of the WGP's 320 KB

// workspace layout (bytes)
#define WS_W        0u
#define WS_W_BYTES  (10u * 512u * 1024u)        // 10 blocks of 512x512 f16 = 5.24 MB
#define WS_H16      (WS_W_BYTES)                // 2 x 163840 B  (f16 h ping-pong)
#define WS_H32      (WS_H16 + 2u * 163840u)     // 2 x 327680 B  (f32 h ping-pong)
#define WS_BAR      (WS_H32 + 2u * 327680u)     // grid barrier {count, generation}

// d_out layout (floats): mean[32000], std[32000], rnn_out[81920000], hn_last[81920]
#define OUT_MEAN 0
#define OUT_STD  32000
#define OUT_RNN  64000
#define OUT_HN   (64000 + 81920000)

__constant__ int c_nblk[5]     = {3, 1, 2, 2, 2};
__constant__ int c_blkid[5][3] = {{0,1,2},{3,0,0},{4,5,0},{6,7,0},{8,9,0}};
__constant__ int c_inreg[5][3] = {{0,3,4},{0,0,0},{0,1,0},{2,4,0},{3,4,0}};

__device__ __forceinline__ float ht(float w) {
    return fminf(fmaxf(w, 1e-10f), 1.0f);
}

// ---------------------------------------------------------------------------
// Build the 10 effective weight blocks (hardtanh, Dale signs, column masks) in
// f16, packed per 16x32 WMMA-B fragment: halves of lane l are contiguous 32 B.
// slot(half hh, lane l) = W_rec[n = 16*n_tile + l%16, j = 32*k_chunk + kt(l/16, hh)], kt = ISA table K order
// ---------------------------------------------------------------------------
__global__ void pack_weights(const float* w_str2snr, const float* w_m12m1,
                             const float* w_m12str,  const float* w_thal2m1,
                             const float* w_m12thal, const float* w_thal2str,
                             const float* w_str2stn, const float* w_snr2thal,
                             const float* str2str_fixed, _Float16* packed) {
    unsigned idx = blockIdx.x * blockDim.x + threadIdx.x;   // one per half
    const unsigned PER_BLK = 512u * 512u;
    if (idx >= 10u * PER_BLK) return;
    int bl = idx / PER_BLK;
    unsigned r  = idx % PER_BLK;
    int nlt  = r / 8192;          // n_loc_tile within region (0..31)
    unsigned r2 = r % 8192;
    int kc   = r2 / 512;          // k_chunk (0..15)
    unsigned r3 = r2 % 512;
    int lane = r3 / 16;           // 0..31 -> WMMA lane (N column = lane%16)
    int hh   = r3 % 16;           // 0..15 -> element slot (K row, table order)
    int i_loc = nlt * 16 + (lane & 15);    // out-neuron within region (block row)
    int j_loc = kc * 32 + ((hh < 8) ? (8 * (lane >> 4) + hh) : (16 + 8 * (lane >> 4) + (hh - 8)));   // in-neuron within region (block col)
    int e = i_loc * 512 + j_loc;
    float v;
    switch (bl) {
        case 0: v = -str2str_fixed[e]; break;                          // str<-str
        case 1: v =  ht(w_thal2str[e]); break;                         // str<-thal
        case 2: v = (j_loc < NEXC) ?  ht(w_m12str[e]) : 0.0f; break;   // str<-m1 (exc cols)
        case 3: v = (j_loc >= 256) ?  ht(w_str2stn[e]) : 0.0f; break;  // stn<-str (2nd half)
        case 4: v = (j_loc < 256)  ? -ht(w_str2snr[e]) : 0.0f; break;  // snr<-str (1st half, -)
        case 5: v = -ht(w_str2stn[e]); break;                          // snr<-stn (faithful reuse)
        case 6: v = -ht(w_snr2thal[e]); break;                         // thal<-snr
        case 7: v =  ht(w_m12thal[e]); break;                          // thal<-m1
        case 8: v =  ht(w_thal2m1[e]); break;                          // m1<-thal
        default: v = (j_loc < NEXC) ? ht(w_m12m1[e]) : -ht(w_m12m1[e]); break; // m1<-m1 Dale
    }
    packed[idx] = (_Float16)v;
}

// ---------------------------------------------------------------------------
__global__ void init_state(const float* hn, _Float16* h16, float* h32, unsigned* bar) {
    int i = blockIdx.x * 256 + threadIdx.x;
    if (i < B_ * HTOT) { float v = hn[i]; h32[i] = v; h16[i] = (_Float16)v; }
    if (blockIdx.x == 0 && threadIdx.x < 2) bar[threadIdx.x] = 0u;  // count, generation
}

// ---------------------------------------------------------------------------
__device__ __forceinline__ void grid_barrier(unsigned* bar) {
    __syncthreads();
    __threadfence();   // make this step's h stores visible at agent scope
    if (threadIdx.x == 0) {
        unsigned g = __hip_atomic_load(&bar[1], __ATOMIC_ACQUIRE, __HIP_MEMORY_SCOPE_AGENT);
        unsigned arrived = __hip_atomic_fetch_add(&bar[0], 1u, __ATOMIC_ACQ_REL,
                                                  __HIP_MEMORY_SCOPE_AGENT);
        if (arrived == NWG - 1) {
            __hip_atomic_store(&bar[0], 0u, __ATOMIC_RELAXED, __HIP_MEMORY_SCOPE_AGENT);
            __hip_atomic_fetch_add(&bar[1], 1u, __ATOMIC_RELEASE, __HIP_MEMORY_SCOPE_AGENT);
        } else {
            while (__hip_atomic_load(&bar[1], __ATOMIC_ACQUIRE,
                                     __HIP_MEMORY_SCOPE_AGENT) == g)
                __builtin_amdgcn_s_sleep(1);
        }
    }
    __syncthreads();
    __threadfence();   // drop stale ping-pong lines from near caches
}

// ---------------------------------------------------------------------------
// Persistent RNN: 40 WGs x 8 waves. Wave = one 16(batch) x 16(neuron) output
// tile; its block-sparse K columns live in LDS for the whole 1000-step run.
// ---------------------------------------------------------------------------
__global__ void __launch_bounds__(256) rnn_persistent(
        const float* __restrict__ inp, const float* __restrict__ inp_weight,
        const _Float16* __restrict__ packedW, _Float16* __restrict__ h16,
        float* __restrict__ h32, unsigned* bar,
        float* __restrict__ out_rnn, float* __restrict__ out_hn) {
    extern __shared__ unsigned char smem[];
    const int tid  = threadIdx.x;
    const int wave = tid >> 5, lane = tid & 31;
    const int slot = wave >> 1, m_t = wave & 1;
    const int n_t  = blockIdx.x * 4 + slot;        // 0..159
    const int r_out = n_t >> 5;                    // out region 0..4
    const int nblk  = c_nblk[r_out];

    // ---- one-time stage of this WG's weight slices into LDS (contiguous 16KB each)
    for (int s = 0; s < 4; ++s) {
        int nt_s = blockIdx.x * 4 + s;
        int rs = nt_s >> 5, nlt = nt_s & 31;
        for (int bi = 0; bi < c_nblk[rs]; ++bi) {
            const float4* src = (const float4*)(packedW
                + (size_t)c_blkid[rs][bi] * 262144u + (size_t)nlt * 8192u);
            float4* dst = (float4*)(smem + s * SLOT_LDS + bi * BLK_SLICE);
            for (int e = tid; e < 1024; e += 256) dst[e] = src[e];
        }
    }
    __syncthreads();

    const int ncol = (n_t << 4) + (lane & 15);     // global out-neuron of D column
    float w0 = 0.f, w1 = 0.f, w2 = 0.f, w3 = 0.f;  // hoisted input projection (str only)
    if (r_out == 0) {
        w0 = 0.3f * ht(inp_weight[0 * HTOT + ncol]);
        w1 = 0.3f * ht(inp_weight[1 * HTOT + ncol]);
        w2 = 0.3f * ht(inp_weight[2 * HTOT + ncol]);
        w3 = 0.3f * ht(inp_weight[3 * HTOT + ncol]);
    }
    const float tonic = (r_out >= 1 && r_out <= 3) ? 0.1f : 0.0f;
    const int arow = m_t * 16 + (lane & 15);       // batch row this lane loads for A
    const int khi  = (lane >= 16) ? 8 : 0;         // ISA A-layout: hi half-wave K offset
    const int bbase = m_t * 16 + ((lane >= 16) ? 8 : 0);

    for (int t = 0; t < T_; ++t) {
        const _Float16* hp16 = h16 + (size_t)(t & 1) * (B_ * HTOT);
        _Float16*       hx16 = h16 + (size_t)((t + 1) & 1) * (B_ * HTOT);
        const float*    hp32 = h32 + (size_t)(t & 1) * (B_ * HTOT);
        float*          hx32 = h32 + (size_t)((t + 1) & 1) * (B_ * HTOT);

        const _Float16* arow_p = hp16 + (size_t)arow * HTOT;
        __builtin_prefetch(arow_p, 0, 0);          // global_prefetch_b8 of h row

        v8f acc = {0.f, 0.f, 0.f, 0.f, 0.f, 0.f, 0.f, 0.f};
        for (int bi = 0; bi < nblk; ++bi) {
            const int kgbase = c_inreg[r_out][bi] * HIDD;
            const _Float16* lbase =
                (const _Float16*)(smem + slot * SLOT_LDS + bi * BLK_SLICE);
            #pragma unroll 4
            for (int c = 0; c < 16; ++c) {
                // A fragment (16x32 f16): two contiguous 16B runs per lane (ISA layout)
                const _Float16* ap = arow_p + kgbase + c * 32 + khi;
                v8h alo = *(const v8h*)(ap);
                v8h ahi = *(const v8h*)(ap + 16);
                v16h a = __builtin_shufflevector(alo, ahi,
                    0,1,2,3,4,5,6,7,8,9,10,11,12,13,14,15);
                // B fragment from LDS: lane's 16 halves contiguous (2x ds_load_b128)
                const _Float16* bp = lbase + (c * 32 + lane) * 16;
                v8h blo = *(const v8h*)(bp);
                v8h bhi = *(const v8h*)(bp + 8);
                v16h bf = __builtin_shufflevector(blo, bhi,
                    0,1,2,3,4,5,6,7,8,9,10,11,12,13,14,15);
                acc = __builtin_amdgcn_wmma_f32_16x16x32_f16(
                    false, a, false, bf, (short)0, acc, false, false);
            }
        }

        // ---- elementwise update + stores (D layout: VGPR r -> batch row bbase+r)
        #pragma unroll
        for (int r = 0; r < 8; ++r) {
            const int b = bbase + r;
            float x = 0.0f;
            if (r_out == 0) {
                float4 i4 = *(const float4*)(inp + ((size_t)b * T_ + t) * INP_);
                x = i4.x * w0 + i4.y * w1 + i4.z * w2 + i4.w * w3;
            }
            float hprev = hp32[(size_t)b * HTOT + ncol];
            float val = (1.0f - TAUF) * hprev + TAUF * (acc[r] + x) + tonic;
            float hne = fmaxf(val, 0.0f);
            hx32[(size_t)b * HTOT + ncol] = hne;
            hx16[(size_t)b * HTOT + ncol] = (_Float16)hne;
            out_rnn[((size_t)b * T_ + t) * HTOT + ncol] = hne;
            if (t == T_ - 1) out_hn[(size_t)b * HTOT + ncol] = hne;
        }
        grid_barrier(bar);
    }
}

// ---------------------------------------------------------------------------
// mean/std head: one wave per (b,t); only the m1 region (last 512) contributes.
// ---------------------------------------------------------------------------
__global__ void head_kernel(const float* __restrict__ rnn,
                            const float* __restrict__ mean_w, const float* __restrict__ mean_b,
                            const float* __restrict__ std_w,  const float* __restrict__ std_b,
                            float* __restrict__ out) {
    int gw = (blockIdx.x * blockDim.x + threadIdx.x) >> 5;
    int lane = threadIdx.x & 31;
    if (gw >= B_ * T_) return;
    int b = gw / T_, t = gw % T_;
    const float* row = rnn + ((size_t)b * T_ + t) * HTOT + 4 * HIDD;
    float sm = 0.f, ss = 0.f;
    for (int j = lane; j < HIDD; j += 32) {
        float v = row[j];
        sm += v * mean_w[4 * HIDD + j];
        ss += v * std_w[4 * HIDD + j];
    }
    for (int o = 16; o > 0; o >>= 1) { sm += __shfl_xor(sm, o); ss += __shfl_xor(ss, o); }
    if (lane == 0) {
        out[OUT_MEAN + gw] = sm + mean_b[0];
        out[OUT_STD  + gw] = fminf(fmaxf(ss + std_b[0], -20.0f), 10.0f);
    }
}

// ---------------------------------------------------------------------------
extern "C" void kernel_launch(void* const* d_in, const int* in_sizes, int n_in,
                              void* d_out, int out_size, void* d_ws, size_t ws_size,
                              hipStream_t stream) {
    (void)in_sizes; (void)n_in; (void)out_size; (void)ws_size;
    const float* inp        = (const float*)d_in[0];
    const float* hn         = (const float*)d_in[1];
    const float* w_str2snr  = (const float*)d_in[3];
    const float* w_m12m1    = (const float*)d_in[4];
    const float* w_m12str   = (const float*)d_in[5];
    const float* w_thal2m1  = (const float*)d_in[6];
    const float* w_m12thal  = (const float*)d_in[7];
    const float* w_thal2str = (const float*)d_in[8];
    const float* w_str2stn  = (const float*)d_in[9];
    const float* w_snr2thal = (const float*)d_in[11];
    const float* s2s_fixed  = (const float*)d_in[12];
    const float* inp_weight = (const float*)d_in[13];
    const float* mean_w     = (const float*)d_in[14];
    const float* mean_b     = (const float*)d_in[15];
    const float* std_w      = (const float*)d_in[16];
    const float* std_b      = (const float*)d_in[17];

    float* out = (float*)d_out;
    unsigned char* ws = (unsigned char*)d_ws;
    _Float16* packedW = (_Float16*)(ws + WS_W);
    _Float16* h16     = (_Float16*)(ws + WS_H16);
    float*    h32     = (float*)(ws + WS_H32);
    unsigned* bar     = (unsigned*)(ws + WS_BAR);

    pack_weights<<<10240, 256, 0, stream>>>(w_str2snr, w_m12m1, w_m12str, w_thal2m1,
                                            w_m12thal, w_thal2str, w_str2stn,
                                            w_snr2thal, s2s_fixed, packedW);
    init_state<<<(B_ * HTOT + 255) / 256, 256, 0, stream>>>(hn, h16, h32, bar);
    rnn_persistent<<<NWG, 256, LDS_BYTES, stream>>>(inp, inp_weight, packedW, h16, h32,
                                                    bar, out + OUT_RNN, out + OUT_HN);
    head_kernel<<<(B_ * T_ * 32 + 255) / 256, 256, 0, stream>>>(out + OUT_RNN, mean_w,
                                                                mean_b, std_w, std_b, out);
}
